// ThalamicAttention_80882824118916
// MI455X (gfx1250) — matrix operand order repaired and checked by reading, not compiled, not run on hardware
//
#include <hip/hip_runtime.h>

typedef __bf16 bf16;
typedef __attribute__((ext_vector_type(16))) bf16 v16bf;
typedef __attribute__((ext_vector_type(8))) float v8f;
typedef __attribute__((ext_vector_type(4))) unsigned int u32x4;

#define B_ 2
#define S_ 2048
#define D_ 1024
#define H_ 16
#define HD_ 64
#define KPQ_ 204   // max(1, int(2048 * 0.1))

__device__ __forceinline__ bf16 f2bf(float f) {
  unsigned u = __float_as_uint(f);
  unsigned r = (u + 0x7FFFu + ((u >> 16) & 1u)) >> 16;   // round-to-nearest-even
  unsigned short h = (unsigned short)r;
  bf16 out;
  __builtin_memcpy(&out, &h, 2);
  return out;
}

union AFrag { v16bf v; u32x4 q[2]; };

__device__ __forceinline__ v8f zero8() {
  v8f z;
#pragma unroll
  for (int i = 0; i < 8; ++i) z[i] = 0.0f;
  return z;
}

__device__ __forceinline__ float wredmax(float v) {
#pragma unroll
  for (int m = 16; m >= 1; m >>= 1) v = fmaxf(v, __shfl_xor(v, m, 32));
  return v;
}
__device__ __forceinline__ float wredmin(float v) {
#pragma unroll
  for (int m = 16; m >= 1; m >>= 1) v = fminf(v, __shfl_xor(v, m, 32));
  return v;
}
__device__ __forceinline__ float wredsum(float v) {
#pragma unroll
  for (int m = 16; m >= 1; m >>= 1) v += __shfl_xor(v, m, 32);
  return v;
}
__device__ __forceinline__ int wredsumi(int v) {
#pragma unroll
  for (int m = 16; m >= 1; m >>= 1) v += __shfl_xor(v, m, 32);
  return v;
}

// CDNA5 async global->LDS copy (16B per lane), tracked by ASYNCcnt.
// No "memory" clobber: copies target the *other* LDS buffer, so surrounding
// ds_loads may be freely scheduled around them. Ordering vs the barrier is
// enforced by wait_async0() (which does clobber memory) + volatile asm order.
__device__ __forceinline__ void async_copy16(unsigned lds_off, const void* gptr) {
  asm volatile("global_load_async_to_lds_b128 %0, %1, off"
               :: "v"(lds_off), "v"(gptr));
}
__device__ __forceinline__ void wait_async0() {
  asm volatile("s_wait_asynccnt 0x0" ::: "memory");
}
__device__ __forceinline__ unsigned lds_addr32(const void* p) {
  return (unsigned)(uintptr_t)p;   // generic LDS addr = aperture_hi | offset_lo32
}

// ---------------------------------------------------------------------------
// f32 -> bf16 cast (vectorized x4)
// ---------------------------------------------------------------------------
__global__ void ta_cast(const float* __restrict__ in, bf16* __restrict__ out, int n4) {
  int i = blockIdx.x * blockDim.x + threadIdx.x;
  if (i < n4) {
    const float4 f = ((const float4*)in)[i];
    union { bf16 e[4]; unsigned long long u; } p;
    p.e[0] = f2bf(f.x); p.e[1] = f2bf(f.y); p.e[2] = f2bf(f.z); p.e[3] = f2bf(f.w);
    ((unsigned long long*)out)[i] = p.u;
  }
}

// ---------------------------------------------------------------------------
// GEMM:  C[M=4096][N=1024] = A[M][K=1024] @ W[N][K]^T + bias[N]
// block tile 64(M) x 256(N), 8 waves, each wave 32x64 (2x4 WMMA subtiles).
// Double-buffered LDS staging via async global->LDS copies (ASYNCcnt).
// K-loop fully unrolled so accumulators coalesce in place (no phi copies).
// mode 0: store bf16 at [b][h][s][hd]   (Q, K projections)
// mode 1: store bf16 at [b][h][hd][s]   (V projection, transposed for PV GEMM)
// mode 2: store f32 at [row][col]       (final output projection)
// ---------------------------------------------------------------------------
__global__ __launch_bounds__(256)
void ta_gemm(const bf16* __restrict__ A, const bf16* __restrict__ W,
             const float* __restrict__ bias, void* __restrict__ Out, int mode)
{
  __shared__ bf16 As[2][64 * 32];    // 2 x 4 KB
  __shared__ bf16 Bs[2][256 * 32];   // 2 x 16 KB

  const int tid   = threadIdx.x;
  const int lane  = tid & 31;
  const int wave  = tid >> 5;
  const int mBlock = blockIdx.x * 64;
  const int nBlock = blockIdx.y * 256;
  const int mWave  = (wave >> 2) * 32;
  const int nWave  = (wave & 3) * 64;
  const int fm   = lane & 15;
  const int aoff = (lane < 16) ? 0 : 8;    // A-fragment K group offset
  const int boff = (lane < 16) ? 0 : 8;    // B-fragment K group offset
  const int hi8  = (lane < 16) ? 0 : 8;    // C-fragment row offset

  // per-thread staging coordinates
  const int ar = tid >> 2, ac = (tid & 3) * 8;   // A: one 16B vector / thread

  v8f acc[2][4];
#pragma unroll
  for (int i = 0; i < 2; ++i)
#pragma unroll
    for (int j = 0; j < 4; ++j) acc[i][j] = zero8();

  // stage one K-chunk (32 wide) into buffer `buf` with async copies
  auto stage = [&](int kc, int buf) {
    async_copy16(lds_addr32(&As[buf][ar * 32 + ac]),
                 &A[(size_t)(mBlock + ar) * D_ + kc * 32 + ac]);
#pragma unroll
    for (int j = 0; j < 4; ++j) {
      int vi = tid + 256 * j, r = vi >> 2, c = (vi & 3) * 8;
      async_copy16(lds_addr32(&Bs[buf][r * 32 + c]),
                   &W[(size_t)(nBlock + r) * D_ + kc * 32 + c]);
    }
  };

  stage(0, 0);
  wait_async0();
  __syncthreads();

  const int nK = D_ / 32;
#pragma unroll
  for (int kc = 0; kc < nK; ++kc) {
    const int cur = kc & 1;
    if (kc + 1 < nK) stage(kc + 1, cur ^ 1);   // prefetch overlaps WMMAs below

    AFrag af[2];
#pragma unroll
    for (int ms = 0; ms < 2; ++ms) {
      int row = mWave + ms * 16 + fm;
      af[ms].q[0] = *(const u32x4*)&As[cur][row * 32 + aoff];
      af[ms].q[1] = *(const u32x4*)&As[cur][row * 32 + aoff + 16];
    }
#pragma unroll
    for (int ns = 0; ns < 4; ++ns) {
      AFrag bfg;
      int row = nWave + ns * 16 + fm;
      bfg.q[0] = *(const u32x4*)&Bs[cur][row * 32 + boff];
      bfg.q[1] = *(const u32x4*)&Bs[cur][row * 32 + boff + 16];
#pragma unroll
      for (int ms = 0; ms < 2; ++ms)
        acc[ms][ns] = __builtin_amdgcn_wmma_f32_16x16x32_bf16(
            false, af[ms].v, false, bfg.v, (short)0, acc[ms][ns], false, false);
    }

    wait_async0();     // next chunk resident
    __syncthreads();   // all waves done reading `cur` before it is overwritten
  }

  // epilogue: bias + layout-specific store
#pragma unroll
  for (int ms = 0; ms < 2; ++ms)
#pragma unroll
    for (int ns = 0; ns < 4; ++ns)
#pragma unroll
      for (int r = 0; r < 8; ++r) {
        int gRow = mBlock + mWave + ms * 16 + r + hi8;
        int gCol = nBlock + nWave + ns * 16 + fm;
        float v = acc[ms][ns][r] + bias[gCol];
        int b = gRow / S_, s = gRow % S_;
        int h = gCol / HD_, n = gCol % HD_;
        if (mode == 0) {
          ((bf16*)Out)[(((size_t)(b * H_ + h)) * S_ + s) * HD_ + n] = f2bf(v);
        } else if (mode == 1) {
          ((bf16*)Out)[(((size_t)(b * H_ + h)) * HD_ + n) * S_ + s] = f2bf(v);
        } else {
          ((float*)Out)[(size_t)gRow * D_ + gCol] = v;
        }
      }
}

// ---------------------------------------------------------------------------
// Attention: per block, 16 query rows of one (b,h).
// Phase 1: scores = scale * Q Kt   (WMMA, scores kept in LDS, 128 KB)
// Phase 2: per-row top-204 threshold (binary search, row cached in VGPRs)
//          + softmax; probabilities written back in place as bf16
// Phase 3: O = P @ V               (WMMA, P read as bf16 from LDS)
// 128 threads (4 waves).
// ---------------------------------------------------------------------------
__global__ __launch_bounds__(128)
void ta_attn(const bf16* __restrict__ Qp, const bf16* __restrict__ Kp,
             const bf16* __restrict__ Vt, bf16* __restrict__ Oa)
{
  __shared__ float sc[16 * 2048];   // 128 KB score strip

  const int tid  = threadIdx.x;
  const int lane = tid & 31;
  const int wave = tid >> 5;
  const int nQB  = S_ / 16;
  const int qb = blockIdx.x % nQB;
  const int bh = blockIdx.x / nQB;
  const int b = bh / H_, h = bh % H_;
  const int qBase = qb * 16;

  const int fm   = lane & 15;
  const int aoff = (lane < 16) ? 0 : 8;
  const int boff = (lane < 16) ? 0 : 8;
  const int hi8  = (lane < 16) ? 0 : 8;
  const float scale = 0.125f;   // 1/sqrt(64)

  const bf16* Qb = Qp + (((size_t)bh) * S_ + qBase) * HD_;
  const bf16* Kb = Kp + ((size_t)bh) * S_ * HD_;
  const bf16* Vb = Vt + ((size_t)bh) * HD_ * S_;

  // ---- phase 1: scores -------------------------------------------------
  AFrag aq[2];
#pragma unroll
  for (int kk = 0; kk < 2; ++kk) {
    aq[kk].q[0] = *(const u32x4*)(Qb + fm * HD_ + kk * 32 + aoff);
    aq[kk].q[1] = *(const u32x4*)(Qb + fm * HD_ + kk * 32 + aoff + 16);
  }
#pragma unroll
  for (int ct = 0; ct < 32; ++ct) {              // each wave: 512 key columns
    int col0 = wave * 512 + ct * 16;
    v8f acc = zero8();
#pragma unroll
    for (int kk = 0; kk < 2; ++kk) {
      AFrag bk;
      const bf16* p = Kb + (size_t)(col0 + fm) * HD_ + kk * 32 + boff;
      bk.q[0] = *(const u32x4*)(p);
      bk.q[1] = *(const u32x4*)(p + 16);
      acc = __builtin_amdgcn_wmma_f32_16x16x32_bf16(
          false, aq[kk].v, false, bk.v, (short)0, acc, false, false);
    }
#pragma unroll
    for (int r = 0; r < 8; ++r)
      sc[(r + hi8) * 2048 + col0 + fm] = acc[r] * scale;
  }
  __syncthreads();

  // ---- phase 2: top-k threshold + softmax (each wave: 4 rows) ----------
  // Row cached in 64 VGPRs per lane; binary search is pure VALU.
#pragma unroll 1
  for (int rr = 0; rr < 4; ++rr) {
    int row = wave * 4 + rr;
    float* srow = &sc[row * 2048];
    float rv[64];
#pragma unroll
    for (int j = 0; j < 64; ++j) rv[j] = srow[lane + 32 * j];

    float mx = -3.0e38f, mn = 3.0e38f;
#pragma unroll
    for (int j = 0; j < 64; ++j) { mx = fmaxf(mx, rv[j]); mn = fminf(mn, rv[j]); }
    mx = wredmax(mx);
    mn = wredmin(mn);

    float lo = mn, hi = mx + 1.0f;
    for (int it = 0; it < 22; ++it) {            // locate k-th largest value
      float mid = 0.5f * (lo + hi);
      int cnt = 0;
#pragma unroll
      for (int j = 0; j < 64; ++j) cnt += (rv[j] >= mid) ? 1 : 0;
      cnt = wredsumi(cnt);
      if (cnt >= KPQ_) lo = mid; else hi = mid;
    }
    float thr = 3.0e38f;                         // exact k-th largest element
#pragma unroll
    for (int j = 0; j < 64; ++j) if (rv[j] >= lo) thr = fminf(thr, rv[j]);
    thr = wredmin(thr);

    float sum = 0.0f;
#pragma unroll
    for (int j = 0; j < 64; ++j) {
      float e = (rv[j] >= thr) ? __expf(rv[j] - mx) : 0.0f;
      rv[j] = e;
      sum += e;
    }
    sum = wredsum(sum);
    float inv = 1.0f / sum;

    // write normalized probabilities back IN PLACE as bf16 (row-local alias)
    bf16* prow = (bf16*)srow;
#pragma unroll
    for (int j = 0; j < 64; ++j) prow[lane + 32 * j] = f2bf(rv[j] * inv);
  }
  __syncthreads();

  // ---- phase 3: O = P @ V (each wave owns one 16x16 output tile) -------
  int nBase = wave * 16;
  v8f acc = zero8();
  const bf16* prow = (const bf16*)&sc[fm * 2048];
#pragma unroll
  for (int kc = 0; kc < S_ / 32; ++kc) {
    AFrag ap;
    ap.q[0] = *(const u32x4*)(prow + kc * 32 + aoff);
    ap.q[1] = *(const u32x4*)(prow + kc * 32 + aoff + 16);
    AFrag bv;
    const bf16* p = Vb + (size_t)(nBase + fm) * S_ + kc * 32 + boff;
    bv.q[0] = *(const u32x4*)(p);
    bv.q[1] = *(const u32x4*)(p + 16);
    acc = __builtin_amdgcn_wmma_f32_16x16x32_bf16(
        false, ap.v, false, bv.v, (short)0, acc, false, false);
  }
#pragma unroll
  for (int r = 0; r < 8; ++r) {
    size_t orow = (size_t)b * S_ + qBase + r + hi8;
    Oa[orow * D_ + h * HD_ + nBase + fm] = f2bf(acc[r]);
  }
}

// ---------------------------------------------------------------------------
extern "C" void kernel_launch(void* const* d_in, const int* in_sizes, int n_in,
                              void* d_out, int out_size, void* d_ws, size_t ws_size,
                              hipStream_t stream) {
  (void)in_sizes; (void)n_in; (void)out_size; (void)ws_size;
  const float* q  = (const float*)d_in[0];
  const float* k  = (const float*)d_in[1];
  const float* v  = (const float*)d_in[2];
  const float* Wq = (const float*)d_in[3];
  const float* bq = (const float*)d_in[4];
  const float* Wk = (const float*)d_in[5];
  const float* bk = (const float*)d_in[6];
  const float* Wv = (const float*)d_in[7];
  const float* bv = (const float*)d_in[8];
  const float* Wo = (const float*)d_in[9];
  const float* bo = (const float*)d_in[10];

  char* ws = (char*)d_ws;
  const size_t SZ_IN = (size_t)B_ * S_ * D_ * sizeof(bf16);  // 8 MiB
  const size_t SZ_W  = (size_t)D_ * D_ * sizeof(bf16);       // 2 MiB
  bf16* qx = (bf16*)(ws);
  bf16* kx = (bf16*)(ws + SZ_IN);
  bf16* vx = (bf16*)(ws + 2 * SZ_IN);
  bf16* wq = (bf16*)(ws + 3 * SZ_IN);
  bf16* wk = (bf16*)(ws + 3 * SZ_IN + SZ_W);
  bf16* wv = (bf16*)(ws + 3 * SZ_IN + 2 * SZ_W);
  bf16* wo = (bf16*)(ws + 3 * SZ_IN + 3 * SZ_W);
  bf16* Qp = (bf16*)(ws + 3 * SZ_IN + 4 * SZ_W);
  bf16* Kp = (bf16*)(ws + 4 * SZ_IN + 4 * SZ_W);
  bf16* Vt = (bf16*)(ws + 5 * SZ_IN + 4 * SZ_W);
  bf16* aO = (bf16*)(ws + 6 * SZ_IN + 4 * SZ_W);

  const int nin = B_ * S_ * D_;   // 4194304
  const int nw  = D_ * D_;        // 1048576
  ta_cast<<<(nin / 4 + 255) / 256, 256, 0, stream>>>(q, qx, nin / 4);
  ta_cast<<<(nin / 4 + 255) / 256, 256, 0, stream>>>(k, kx, nin / 4);
  ta_cast<<<(nin / 4 + 255) / 256, 256, 0, stream>>>(v, vx, nin / 4);
  ta_cast<<<(nw / 4 + 255) / 256, 256, 0, stream>>>(Wq, wq, nw / 4);
  ta_cast<<<(nw / 4 + 255) / 256, 256, 0, stream>>>(Wk, wk, nw / 4);
  ta_cast<<<(nw / 4 + 255) / 256, 256, 0, stream>>>(Wv, wv, nw / 4);
  ta_cast<<<(nw / 4 + 255) / 256, 256, 0, stream>>>(Wo, wo, nw / 4);

  dim3 gg(64, 4);   // M/64 x N/256
  ta_gemm<<<gg, 256, 0, stream>>>(qx, wq, bq, (void*)Qp, 0);
  ta_gemm<<<gg, 256, 0, stream>>>(kx, wk, bk, (void*)Kp, 0);
  ta_gemm<<<gg, 256, 0, stream>>>(vx, wv, bv, (void*)Vt, 1);

  ta_attn<<<B_ * H_ * (S_ / 16), 128, 0, stream>>>(Qp, Kp, Vt, aO);

  ta_gemm<<<gg, 256, 0, stream>>>(aO, wo, bo, d_out, 2);
}
